// DGIN_28363964023321
// MI455X (gfx1250) — compile-verified, matrix operand order checked by reading, not run on hardware
//
#include <hip/hip_runtime.h>
#include <hip/hip_bf16.h>

// ---- WMMA types (CDNA5 gfx1250, wave32) ----
typedef __attribute__((ext_vector_type(16))) __bf16 v16bf;
typedef __attribute__((ext_vector_type(8)))  float  v8f;

#define N_NODES 50000
#define N_EDGES 640000

// Fragment index mapping for v_wmma_f32_16x16x32_bf16 (ISA 7.12.2):
//   element e of lane l maps to K = kt*32 + (e/8)*16 + (l/16)*8 + (e%8)
//   A: row M = l%16 ; B: col N = l%16
//   C/D: col N = l%16, row M = v + 8*(l/16) for acc element v.
//
// Weights are pre-packed fragment-major:
//   P[ ((kt*NT + nt)*32 + lane)*16 + e ] = W[K][N]   (zero-padded past Krows)
// so a B-fragment load is one 32-byte coalesced vector load per lane.

__device__ __forceinline__ v16bf load_packed_frag(const __bf16* __restrict__ P,
                                                  int NT, int nt, int kt, int lane) {
  return *(const v16bf*)(P + (((size_t)(kt * NT + nt) * 32 + lane) << 4));
}

// ---- pack one weight matrix f32[Krows x ld] -> bf16 fragment-major ----
__global__ void pack_weights(const float* __restrict__ Wf, __bf16* __restrict__ P,
                             int Krows, int ld, int NT, int total) {
  int i = blockIdx.x * blockDim.x + threadIdx.x;
  if (i >= total) return;
  int e    = i & 15;
  int lane = (i >> 4) & 31;
  int frag = i >> 9;
  int nt = frag % NT;
  int kt = frag / NT;
  int K = kt * 32 + ((e >> 3) << 4) + ((lane >> 4) << 3) + (e & 7);
  int N = nt * 16 + (lane & 15);
  P[i] = (K < Krows) ? (__bf16)Wf[(size_t)K * ld + N] : (__bf16)0.0f;
}

// ---- h = relu(concat(node[src], edge_feat) @ W_init + b_init) ; [E,144]@[144,128]
__global__ __launch_bounds__(128) void edge_init_gemm(
    const float* __restrict__ nf, const float* __restrict__ ef,
    const int* __restrict__ esrc,
    const __bf16* __restrict__ Wi, const float* __restrict__ bi,
    __bf16* __restrict__ h, int ntiles) {
  int wave = blockIdx.x * 4 + (threadIdx.x >> 5);
  if (wave >= ntiles) return;          // wave-uniform: EXEC stays all-ones
  int lane = threadIdx.x & 31;
  int m0 = wave * 16;
  int r  = m0 + (lane & 15);
  int srcr = esrc[r];
  const float* np = nf + (size_t)srcr * 128;
  const float* ep = ef + (size_t)r * 16;
  int half = (lane >> 4) << 3;         // 0 or 8

  v16bf afrag[5];                      // K padded to 160 = 5 x 32
#pragma unroll
  for (int kt = 0; kt < 4; ++kt) {     // node-feature part: K in [0,128)
    int kb = kt * 32 + half;
#pragma unroll
    for (int e = 0; e < 8; ++e) {
      afrag[kt][e]     = (__bf16)np[kb + e];
      afrag[kt][e + 8] = (__bf16)np[kb + 16 + e];
    }
  }
  {                                    // tail: K in [128,144) = edge feat, [144,160) = 0
#pragma unroll
    for (int e = 0; e < 8; ++e) {
      afrag[4][e]     = (__bf16)ep[half + e];   // K = 128 + half + e  (always < 144)
      afrag[4][e + 8] = (__bf16)0.0f;           // K = 144 + half + e  (pad)
    }
  }
#pragma unroll
  for (int nt = 0; nt < 8; ++nt) {
    v8f acc = {};
#pragma unroll
    for (int kt = 0; kt < 5; ++kt) {
      v16bf bf = load_packed_frag(Wi, 8, nt, kt, lane);
      acc = __builtin_amdgcn_wmma_f32_16x16x32_bf16(false, afrag[kt], false, bf,
                                                    (short)0, acc, false, false);
    }
    int n = nt * 16 + (lane & 15);
    float bn = bi[n];
#pragma unroll
    for (int v = 0; v < 8; ++v) {
      int m = m0 + v + half;
      h[(size_t)m * 128 + n] = (__bf16)fmaxf(acc[v] + bn, 0.0f);
    }
  }
}

// ---- h_out = relu((agg[src]-h[rev]) @ W_upd + b + h) ; [E,128]@[128,128]
__global__ __launch_bounds__(128) void edge_update_gemm(
    const float* __restrict__ agg, const __bf16* __restrict__ h,
    const int* __restrict__ esrc, const int* __restrict__ erev,
    const __bf16* __restrict__ Wu, const float* __restrict__ bu,
    __bf16* __restrict__ hout, int ntiles) {
  int wave = blockIdx.x * 4 + (threadIdx.x >> 5);
  if (wave >= ntiles) return;
  int lane = threadIdx.x & 31;
  int m0 = wave * 16;
  int r  = m0 + (lane & 15);
  int srcr = esrc[r];
  int revr = erev[r];
  const float*  ag = agg + (size_t)srcr * 128;
  const __bf16* hr = h   + (size_t)revr * 128;
  int half = (lane >> 4) << 3;

  v16bf afrag[4];                      // K = 128 = 4 x 32 (message in bf16)
#pragma unroll
  for (int kt = 0; kt < 4; ++kt) {
    int kb = kt * 32 + half;
#pragma unroll
    for (int e = 0; e < 8; ++e) {
      afrag[kt][e]     = (__bf16)(ag[kb + e]      - (float)hr[kb + e]);
      afrag[kt][e + 8] = (__bf16)(ag[kb + 16 + e] - (float)hr[kb + 16 + e]);
    }
  }
#pragma unroll
  for (int nt = 0; nt < 8; ++nt) {
    v8f acc = {};
#pragma unroll
    for (int kt = 0; kt < 4; ++kt) {
      v16bf bf = load_packed_frag(Wu, 8, nt, kt, lane);
      acc = __builtin_amdgcn_wmma_f32_16x16x32_bf16(false, afrag[kt], false, bf,
                                                    (short)0, acc, false, false);
    }
    int n = nt * 16 + (lane & 15);
    float bn = bu[n];
#pragma unroll
    for (int v = 0; v < 8; ++v) {
      int m = m0 + v + half;
      float val = acc[v] + bn + (float)h[(size_t)m * 128 + n];   // residual
      hout[(size_t)m * 128 + n] = (__bf16)fmaxf(val, 0.0f);
    }
  }
}

// ---- x0 = (xcat*(1+eps)) @ W_node + b_node ; [N,256]@[256,256]
__global__ __launch_bounds__(128) void node_gemm(
    const float* __restrict__ x, const __bf16* __restrict__ Wn,
    const float* __restrict__ bn, const float* __restrict__ eps_p,
    float* __restrict__ x0, int ntiles) {
  int wave = blockIdx.x * 4 + (threadIdx.x >> 5);
  if (wave >= ntiles) return;
  int lane = threadIdx.x & 31;
  int m0 = wave * 16;
  int r  = m0 + (lane & 15);
  float scale = 1.0f + eps_p[0];
  const float* xp = x + (size_t)r * 256;
  int half = (lane >> 4) << 3;

  v16bf afrag[8];                      // K = 256 = 8 x 32
#pragma unroll
  for (int kt = 0; kt < 8; ++kt) {
    int kb = kt * 32 + half;
#pragma unroll
    for (int e = 0; e < 8; ++e) {
      afrag[kt][e]     = (__bf16)xp[kb + e];
      afrag[kt][e + 8] = (__bf16)xp[kb + 16 + e];
    }
  }
#pragma unroll
  for (int nt = 0; nt < 16; ++nt) {
    v8f acc = {};
#pragma unroll
    for (int kt = 0; kt < 8; ++kt) {
      v16bf bf = load_packed_frag(Wn, 16, nt, kt, lane);
      acc = __builtin_amdgcn_wmma_f32_16x16x32_bf16(false, afrag[kt], false, bf,
                                                    (short)0, acc, false, false);
    }
    int n = nt * 16 + (lane & 15);
    float bb = bn[n];
#pragma unroll
    for (int v = 0; v < 8; ++v) {
      int m = m0 + v + half;
      x0[(size_t)m * 256 + n] = acc[v] * scale + bb;
    }
  }
}

// ---- scatter-add h (bf16) into agg (f32), segment_sum over edge_dst ----
__global__ void scatter_edges(const __bf16* __restrict__ h,
                              const int* __restrict__ edst,
                              float* __restrict__ agg) {
  long t = (long)blockIdx.x * blockDim.x + threadIdx.x;   // E*32 threads, 4 dims each
  int e = (int)(t >> 5);
  int q = ((int)t & 31) << 2;
  int dst = edst[e];
  const __bf16* hp = h + (size_t)e * 128 + q;
  float* ap = agg + (size_t)dst * 128 + q;
#pragma unroll
  for (int j = 0; j < 4; ++j) atomicAdd(ap + j, (float)hp[j]);
}

// ---- scatter-add x[src] (f32,256) into xdst ----
__global__ void scatter_nodes(const float* __restrict__ x,
                              const int* __restrict__ esrc,
                              const int* __restrict__ edst,
                              float* __restrict__ xdst) {
  long t = (long)blockIdx.x * blockDim.x + threadIdx.x;   // E*64 threads, 4 dims each
  int e = (int)(t >> 6);
  int q = ((int)t & 63) << 2;
  int s = esrc[e], d = edst[e];
  const float* xp = x + (size_t)s * 256 + q;
  float* op = xdst + (size_t)d * 256 + q;
#pragma unroll
  for (int j = 0; j < 4; ++j) atomicAdd(op + j, xp[j]);
}

__global__ void zero_f32(float* __restrict__ p, long n) {
  long i = (long)blockIdx.x * blockDim.x + threadIdx.x;
  if (i < n) p[i] = 0.0f;
}

__global__ void copy_f32(float* __restrict__ dst, const float* __restrict__ src, long n) {
  long i = (long)blockIdx.x * blockDim.x + threadIdx.x;
  if (i < n) dst[i] = src[i];
}

// ---- xcat = concat(node_feature, agg) ----
__global__ void build_xcat(const float* __restrict__ nf, const float* __restrict__ agg,
                           float* __restrict__ x) {
  long i = (long)blockIdx.x * blockDim.x + threadIdx.x;   // N*256
  int n = (int)(i >> 8);
  int d = (int)i & 255;
  x[i] = (d < 128) ? nf[(size_t)n * 128 + d] : agg[(size_t)n * 128 + (d - 128)];
}

extern "C" void kernel_launch(void* const* d_in, const int* in_sizes, int n_in,
                              void* d_out, int out_size, void* d_ws, size_t ws_size,
                              hipStream_t stream) {
  const float* node_feature = (const float*)d_in[0];
  const float* edge_feature = (const float*)d_in[1];
  const int*   edge_src     = (const int*)d_in[2];
  const int*   edge_dst     = (const int*)d_in[3];
  const int*   rev_edge     = (const int*)d_in[4];
  const float* W_init       = (const float*)d_in[5];
  const float* b_init       = (const float*)d_in[6];
  const float* W_upd        = (const float*)d_in[7];
  const float* b_upd        = (const float*)d_in[8];
  const float* W_node       = (const float*)d_in[9];
  const float* b_node       = (const float*)d_in[10];
  const float* epsilon      = (const float*)d_in[11];

  const int N = N_NODES, E = N_EDGES;

  // workspace carve-out
  char* ws = (char*)d_ws;
  size_t off = 0;
  auto carve = [&](size_t bytes) -> void* {
    void* p = ws + off;
    off = (off + bytes + 255) & ~(size_t)255;
    return p;
  };
  __bf16* hA  = (__bf16*)carve((size_t)E * 128 * 2);   // 164 MB (fits L2)
  __bf16* hB  = (__bf16*)carve((size_t)E * 128 * 2);
  float*  agg = (float*)carve((size_t)N * 128 * 4);    // 25.6 MB
  float*  x0  = (float*)carve((size_t)N * 256 * 4);    // 51.2 MB
  float*  xB  = (float*)carve((size_t)N * 256 * 4);
  __bf16* Wi  = (__bf16*)carve((size_t)5 * 8 * 512 * 2);    // packed fragments
  __bf16* Wu  = (__bf16*)carve((size_t)4 * 8 * 512 * 2);
  __bf16* Wn  = (__bf16*)carve((size_t)8 * 16 * 512 * 2);

  // 1) weights -> bf16, pre-packed into WMMA fragment layout
  pack_weights<<<(5 * 8 * 512 + 255) / 256, 256, 0, stream>>>(W_init, Wi, 144, 128, 8,
                                                              5 * 8 * 512);
  pack_weights<<<(4 * 8 * 512 + 255) / 256, 256, 0, stream>>>(W_upd, Wu, 128, 128, 8,
                                                              4 * 8 * 512);
  pack_weights<<<(8 * 16 * 512 + 255) / 256, 256, 0, stream>>>(W_node, Wn, 256, 256, 16,
                                                               8 * 16 * 512);

  // 2) initial edge projection: 40000 M-tiles, 4 waves/block
  edge_init_gemm<<<10000, 128, 0, stream>>>(node_feature, edge_feature, edge_src,
                                            Wi, b_init, hA, 40000);

  // 3) edge message passing (4 steps), double-buffered h
  __bf16* hc = hA;
  __bf16* hn = hB;
  for (int s = 0; s < 4; ++s) {
    zero_f32<<<(long)N * 128 / 256, 256, 0, stream>>>(agg, (long)N * 128);
    scatter_edges<<<(long)E * 32 / 256, 256, 0, stream>>>(hc, edge_dst, agg);
    edge_update_gemm<<<10000, 128, 0, stream>>>(agg, hc, edge_src, rev_edge,
                                                Wu, b_upd, hn, 40000);
    __bf16* t = hc; hc = hn; hn = t;
  }

  // 4) final aggregation, concat, GIN projection
  zero_f32<<<(long)N * 128 / 256, 256, 0, stream>>>(agg, (long)N * 128);
  scatter_edges<<<(long)E * 32 / 256, 256, 0, stream>>>(hc, edge_dst, agg);
  build_xcat<<<(long)N * 256 / 256, 256, 0, stream>>>(node_feature, agg, (float*)d_out);
  node_gemm<<<(3125 + 3) / 4, 128, 0, stream>>>((const float*)d_out, Wn, b_node,
                                                epsilon, x0, 3125);

  // 5) node message passing (4 steps), ping-pong so final lands in d_out
  float* xcur = (float*)d_out;   // currently holds xcat (step-0 gather source)
  float* xoth = xB;
  for (int s = 0; s < 4; ++s) {
    copy_f32<<<(long)N * 256 / 256, 256, 0, stream>>>(xoth, x0, (long)N * 256);
    scatter_nodes<<<(long)E * 64 / 256, 256, 0, stream>>>(xcur, edge_src, edge_dst, xoth);
    float* t = xcur; xcur = xoth; xoth = t;
  }
  // after 4 steps xcur == d_out (B, A, B, A) -> result is in d_out
}
